// CapsNet_4372276707737
// MI455X (gfx1250) — matrix operand order repaired and checked by reading, not compiled, not run on hardware
//
#include <hip/hip_runtime.h>
#include <cstdint>
#include <cstddef>

typedef __attribute__((ext_vector_type(16))) __bf16 v16bf;
typedef __attribute__((ext_vector_type(8)))  float  v8f;
typedef __attribute__((ext_vector_type(4)))  unsigned int uint32x4;
typedef __attribute__((ext_vector_type(8)))  int    int32x8;
typedef __attribute__((ext_vector_type(4)))  int    int32x4;

// ---------------- problem constants ----------------
#define BATCH      256
#define C1_K       81            // 9*9*1
#define C1_N       (BATCH*400)   // 102400 (20x20 outputs)
#define C2_K       20736         // 256*81
#define C2_N       (BATCH*36)    // 9216 (6x6 outputs)
#define IS         1152
#define NCAP       8
#define NCLS       10
#define OUTC       16

// ---------------- workspace layout (bytes, all 256-aligned) ----------------
#define OFF_XBF    0u                     // 200704 bf16
#define OFF_WCBF   401408u                // 20736  bf16
#define OFF_WPBF   442880u                // 5308416 bf16
#define OFF_HBF    11059712u              // 26214400 bf16  [c][b*400+y*20+x]
#define OFF_P      63488512u              // 2359296 f32    [c][b*36+s]
#define OFF_UI     72925696u              // 2359296 f32    [b][i][k]
#define OFF_UJI    82362880u              // 47185920 f32   [b][i][c][o]
#define OFF_BIJ    271106560u             // 11520 f32
#define OFF_CIJ    271152640u             // 11520 f32
#define OFF_SJ     271198720u             // 40960 f32
#define OFF_OUTW   271362560u             // 40960 f32

// compile-time im2col offset helpers (k -> ky*rowstride + kx for 9x9 window)
static __device__ __forceinline__ constexpr int kwin(int k, int rs) {
    return (k < C1_K) ? (k / 9) * rs + (k % 9) : 0;
}
static __device__ __forceinline__ constexpr int kclamp(int k) {
    return (k < C1_K) ? k : 0;
}

// ---------------- TDM: 2D tile of bf16 -> LDS (CDNA5 ISA, D# groups 0/1) ------
static __device__ __forceinline__ void tdm_load_bf16_2d(
        uint32_t lds_byte_addr, const void* gptr,
        uint32_t tensor_d0, uint32_t tensor_d1,
        uint32_t tile_d0, uint32_t tile_d1, uint64_t stride0_elems) {
    uint32x4 g0;
    int32x8  g1;
    int32x4  gz = {};
    const uint64_t ga = (uint64_t)(uintptr_t)gptr;
    g0[0] = 1u;                                           // count=1, user D#
    g0[1] = lds_byte_addr;                                // lds_addr [63:32]
    g0[2] = (uint32_t)ga;                                 // global_addr lo
    g0[3] = (uint32_t)((ga >> 32) & 0x1FFFFFFu) | (2u << 30);  // ga hi | type=2
    g1[0] = (int)(1u << 16);                              // data_size=1 (2B), mask=0
    g1[1] = (int)((tensor_d0 & 0xFFFFu) << 16);           // tensor_dim0[15:0] @bit48
    g1[2] = (int)(((tensor_d0 >> 16) & 0xFFFFu) | ((tensor_d1 & 0xFFFFu) << 16));
    g1[3] = (int)(((tensor_d1 >> 16) & 0xFFFFu) | (tile_d0 << 16));  // tile_dim0
    g1[4] = (int)(tile_d1 & 0xFFFFu);                     // tile_dim1, tile_dim2=0
    g1[5] = (int)(uint32_t)(stride0_elems & 0xFFFFFFFFu); // tensor_dim0_stride lo
    g1[6] = (int)(uint32_t)((stride0_elems >> 32) & 0xFFFFu);
    g1[7] = 0;
#if __has_include(<hip/amd_detail/amd_gfx1250_TDM.h>)
    int32x8 gz8 = {};
    __builtin_amdgcn_tensor_load_to_lds(g0, g1, gz, gz, gz8, 0);
#else
    __builtin_amdgcn_tensor_load_to_lds(g0, g1, gz, gz, 0);
#endif
}

__global__ void __launch_bounds__(256) cvt_bf16_kernel(const float* __restrict__ src,
                                                       __bf16* __restrict__ dst, int n) {
    int i = blockIdx.x * 256 + threadIdx.x;
    if (i < n) dst[i] = (__bf16)src[i];
}

__global__ void __launch_bounds__(256) zero_f32_kernel(float* __restrict__ dst, int n) {
    int i = blockIdx.x * 256 + threadIdx.x;
    if (i < n) dst[i] = 0.0f;
}

// ---------------- conv1: implicit GEMM, bf16 WMMA, ReLU epilogue ----------------
// A[m,k] = conv_w[m, ky, kx] (k = ky*9+kx, pad to 96)
// B[k,n] = x[b, oy+ky, ox+kx],  n = b*400 + oy*20 + ox
// All fragment offsets are selects between compile-time constants on (lane<16).
__global__ void __launch_bounds__(256) conv1_wmma_kernel(const __bf16* __restrict__ xb,
                                                         const __bf16* __restrict__ wb,
                                                         const float*  __restrict__ bias,
                                                         __bf16* __restrict__ hb) {
    const int lane = threadIdx.x & 31;
    const bool lo  = (lane < 16);
    const int wg   = blockIdx.x * 8 + (threadIdx.x >> 5);
    const int mt   = wg & 15;          // 16 m-tiles
    const int nt   = wg >> 4;          // 6400 n-tiles

    const int m_loc = lane & 15;
    const int n_loc = lane & 15;

    const int n   = nt * 16 + n_loc;
    const int b   = n / 400;
    const int rem = n - b * 400;
    const int oy  = rem / 20;
    const int ox  = rem - oy * 20;
    const __bf16* xbase = xb + (b * 784 + oy * 28 + ox);
    const __bf16* wrow  = wb + (mt * 16 + m_loc) * C1_K;

    v8f acc = {};
    #pragma unroll
    for (int ks = 0; ks < 3; ++ks) {
        const int k0 = ks * 32;
        v16bf a, bf;
        #pragma unroll
        for (int e = 0; e < 8; ++e) {
            const int kl0 = k0 + e,      kh0 = k0 + 8 + e;       // compile-time
            const int kl1 = k0 + 16 + e, kh1 = k0 + 24 + e;
            __bf16 w0 = wrow[lo ? kclamp(kl0) : kclamp(kh0)];
            __bf16 w1 = wrow[lo ? kclamp(kl1) : kclamp(kh1)];
            a[e]     = (lo ? (kl0 < C1_K) : (kh0 < C1_K)) ? w0 : (__bf16)0.0f;
            a[8 + e] = (lo ? (kl1 < C1_K) : (kh1 < C1_K)) ? w1 : (__bf16)0.0f;
        }
        #pragma unroll
        for (int e = 0; e < 16; ++e) {
            const int klo = k0 + e + (e >= 8 ? 8 : 0), khi = klo + 8;   // compile-time
            __bf16 xv = xbase[lo ? kwin(klo, 28) : kwin(khi, 28)];
            bf[e] = (lo ? (klo < C1_K) : (khi < C1_K)) ? xv : (__bf16)0.0f;
        }
        acc = __builtin_amdgcn_wmma_f32_16x16x32_bf16(false, a, false, bf,
                                                      (short)0, acc, false, false);
    }
    #pragma unroll
    for (int r = 0; r < 8; ++r) {
        int m = mt * 16 + r + (lo ? 0 : 8);
        float v = acc[r] + bias[m];
        v = v > 0.0f ? v : 0.0f;
        hb[(size_t)m * C1_N + n] = (__bf16)v;
    }
}

// ---------------- conv2: implicit GEMM over ic, TDM-staged LDS operands ---------
// Per input channel ic: A tile = prim_w[mt*16..+15][ic*81..+80]  (16x81, row stride 20736)
//                       B tile = h[ic][b0*400 .. b0*400+6399]    (16 images, contiguous)
// Both staged by TENSOR_LOAD_TO_LDS, double buffered on TENSORcnt; K padded 81->96.
__global__ void __launch_bounds__(256) conv2_wmma_kernel(const __bf16* __restrict__ wp,
                                                         const __bf16* __restrict__ hb,
                                                         const float*  __restrict__ bias,
                                                         float* __restrict__ p) {
    __shared__ __bf16 Hs[2][16 * 400];   // 25.6 KB
    __shared__ __bf16 As[2][16 * 81];    //  5.2 KB
    const int t    = threadIdx.x;
    const int wave = t >> 5;
    const int lane = t & 31;
    const bool lo  = (lane < 16);
    const int mt   = blockIdx.x & 15;            // 16 m-tiles
    const int ng   = blockIdx.x >> 4;            // 18 column groups (512 cols each)
    const int nt0  = ng * 32 + wave * 4;

    const int m_loc = lane & 15;
    const int n_loc = lane & 15;
    const int b0    = (ng * 512) / 36;           // first batch image staged

    int hloc[4];                                  // LDS im2col bases per n-subtile
    #pragma unroll
    for (int s = 0; s < 4; ++s) {
        int n  = (nt0 + s) * 16 + n_loc;
        int b  = n / 36;
        int sp = n - b * 36;
        int oy = sp / 6;
        int ox = sp - oy * 6;
        hloc[s] = (b - b0) * 400 + oy * 40 + ox * 2;
    }

    const bool issuer = (wave == 0);
    const uint32_t ldsH0 = (uint32_t)(uintptr_t)&Hs[0][0];
    const uint32_t ldsH1 = (uint32_t)(uintptr_t)&Hs[1][0];
    const uint32_t ldsA0 = (uint32_t)(uintptr_t)&As[0][0];
    const uint32_t ldsA1 = (uint32_t)(uintptr_t)&As[1][0];
    const uint32_t hRemain = (uint32_t)(BATCH - b0) * 400;   // OOB rows read as zero
    const __bf16* wbase = wp + (size_t)(mt * 16) * C2_K;

    v8f acc[4] = {};

    if (issuer) {   // preload ic=0 (only wave 0 reaches here)
        tdm_load_bf16_2d(ldsH0, hb + (size_t)b0 * 400, hRemain, 1, 6400, 1, 0);
        tdm_load_bf16_2d(ldsA0, wbase, C2_K, 4096, 81, 16, C2_K);
    }

    for (int ic = 0; ic < 256; ++ic) {
        const int buf = ic & 1;
        if (issuer) {
            if (ic + 1 < 256) {
                tdm_load_bf16_2d(buf ? ldsH0 : ldsH1,
                                 hb + (size_t)(ic + 1) * C1_N + (size_t)b0 * 400,
                                 hRemain, 1, 6400, 1, 0);
                tdm_load_bf16_2d(buf ? ldsA0 : ldsA1,
                                 wbase + (size_t)(ic + 1) * 81,
                                 (uint32_t)(C2_K - (ic + 1) * 81), 4096, 81, 16, C2_K);
                __builtin_amdgcn_s_wait_tensorcnt((short)2);   // older pair complete
            } else {
                __builtin_amdgcn_s_wait_tensorcnt((short)0);
            }
        }
        __syncthreads();   // staged data for `buf` visible to all waves

        const __bf16* arow = &As[buf][m_loc * 81];
        #pragma unroll
        for (int kc = 0; kc < 3; ++kc) {           // kl = 0..95, >=81 zero padded
            const int k0 = kc * 32;
            v16bf a;
            #pragma unroll
            for (int e = 0; e < 8; ++e) {
                const int kl0 = k0 + e,      kh0 = k0 + 8 + e;   // compile-time
                const int kl1 = k0 + 16 + e, kh1 = k0 + 24 + e;
                __bf16 w0 = arow[lo ? kclamp(kl0) : kclamp(kh0)];
                __bf16 w1 = arow[lo ? kclamp(kl1) : kclamp(kh1)];
                a[e]     = (lo ? (kl0 < C1_K) : (kh0 < C1_K)) ? w0 : (__bf16)0.0f;
                a[8 + e] = (lo ? (kl1 < C1_K) : (kh1 < C1_K)) ? w1 : (__bf16)0.0f;
            }
            #pragma unroll
            for (int s = 0; s < 4; ++s) {
                const __bf16* hrow = &Hs[buf][hloc[s]];
                v16bf bf;
                #pragma unroll
                for (int e = 0; e < 16; ++e) {
                    const int klo = k0 + e + (e >= 8 ? 8 : 0), khi = klo + 8;   // compile-time
                    __bf16 v = hrow[lo ? kwin(klo, 20) : kwin(khi, 20)];
                    bf[e] = (lo ? (klo < C1_K) : (khi < C1_K)) ? v : (__bf16)0.0f;
                }
                acc[s] = __builtin_amdgcn_wmma_f32_16x16x32_bf16(false, a, false, bf,
                                                                 (short)0, acc[s], false, false);
            }
        }
        __syncthreads();   // all reads of `buf` done before TDM retargets it
    }

    #pragma unroll
    for (int s = 0; s < 4; ++s) {
        int n = (nt0 + s) * 16 + n_loc;
        #pragma unroll
        for (int r = 0; r < 8; ++r) {
            int m = mt * 16 + r + (lo ? 0 : 8);
            p[(size_t)m * C2_N + n] = acc[s][r] + bias[m];
        }
    }
}

// ---------------- primary squash -> u_i in (b, i, k) layout ----------------
__global__ void __launch_bounds__(256) squash_primary_kernel(const float* __restrict__ p,
                                                             float* __restrict__ ui) {
    __shared__ float red[256];
    const int bc  = blockIdx.x;        // b*8 + cap
    const int b   = bc >> 3;
    const int cap = bc & 7;
    const int t   = threadIdx.x;

    float ss = 0.0f;
    for (int j = t; j < IS; j += 256) {
        int c  = cap * 32 + j / 36;
        int sp = j % 36;
        float v = p[(size_t)c * C2_N + b * 36 + sp];
        ss += v * v;
    }
    red[t] = ss;
    __syncthreads();
    for (int s = 128; s > 0; s >>= 1) {
        if (t < s) red[t] += red[t + s];
        __syncthreads();
    }
    const float ns    = red[0];
    const float scale = sqrtf(ns) + 1e-10f;
    const float fac   = (ns / (1.0f + ns)) / scale;

    for (int j = t; j < IS; j += 256) {
        int c  = cap * 32 + j / 36;
        int sp = j % 36;
        float v = p[(size_t)c * C2_N + b * 36 + sp];
        ui[((size_t)b * IS + j) * NCAP + cap] = v * fac;
    }
}

// ---------------- u_j_i[b,i,c,o] = sum_k W[i,c,o,k] * u_i[b,i,k] ----------------
__global__ void __launch_bounds__(256) uji_kernel(const float* __restrict__ W,
                                                  const float* __restrict__ ui,
                                                  float* __restrict__ uji) {
    size_t idx = (size_t)blockIdx.x * 256 + threadIdx.x;   // ((b*IS+i)*10+c)*16+o
    int o = idx & 15;
    int c = (idx >> 4) % NCLS;
    size_t bi = idx / (NCLS * OUTC);
    size_t i  = bi % IS;
    const float* wrow = W + ((i * NCLS + c) * OUTC + o) * NCAP;
    const float* urow = ui + bi * NCAP;
    float acc = 0.0f;
    #pragma unroll
    for (int k = 0; k < NCAP; ++k) acc += wrow[k] * urow[k];
    uji[idx] = acc;
}

// ---------------- softmax over i (axis 0) of b_ij (1152,10) ----------------
__global__ void __launch_bounds__(256) softmax_kernel(const float* __restrict__ bij,
                                                      float* __restrict__ cij) {
    __shared__ float red[256];
    const int c = blockIdx.x;
    const int t = threadIdx.x;

    float mx = -3.402823466e38f;
    for (int i = t; i < IS; i += 256) mx = fmaxf(mx, bij[i * NCLS + c]);
    red[t] = mx;
    __syncthreads();
    for (int s = 128; s > 0; s >>= 1) {
        if (t < s) red[t] = fmaxf(red[t], red[t + s]);
        __syncthreads();
    }
    mx = red[0];
    __syncthreads();

    float sum = 0.0f;
    for (int i = t; i < IS; i += 256) sum += __expf(bij[i * NCLS + c] - mx);
    red[t] = sum;
    __syncthreads();
    for (int s = 128; s > 0; s >>= 1) {
        if (t < s) red[t] += red[t + s];
        __syncthreads();
    }
    const float inv = 1.0f / red[0];
    for (int i = t; i < IS; i += 256)
        cij[i * NCLS + c] = __expf(bij[i * NCLS + c] - mx) * inv;
}

// ---------------- s_j[b,c,o] = sum_i c_ij[i,c] * u_j_i[b,i,c,o] ----------------
__global__ void __launch_bounds__(256) sj_kernel(const float* __restrict__ cij,
                                                 const float* __restrict__ uji,
                                                 float* __restrict__ sj) {
    int idx = blockIdx.x * 256 + threadIdx.x;   // (b*10+c)*16+o
    if (idx >= BATCH * NCLS * OUTC) return;
    int o = idx & 15;
    int c = (idx >> 4) % NCLS;
    int b = idx / (NCLS * OUTC);
    const float* up = uji + (((size_t)b * IS) * NCLS + c) * OUTC + o;
    float acc = 0.0f;
    for (int i = 0; i < IS; ++i)
        acc += cij[i * NCLS + c] * up[(size_t)i * NCLS * OUTC];
    sj[idx] = acc;
}

// ---------------- out = squash(s_j, axis=1)  (over the CLASS axis) ----------------
__global__ void __launch_bounds__(256) squash_out_kernel(const float* __restrict__ sj,
                                                         float* __restrict__ out) {
    int idx = blockIdx.x * 256 + threadIdx.x;   // b*16 + o
    if (idx >= BATCH * OUTC) return;
    int b = idx >> 4;
    int o = idx & 15;
    float ns = 0.0f;
    #pragma unroll
    for (int c = 0; c < NCLS; ++c) {
        float v = sj[(b * NCLS + c) * OUTC + o];
        ns += v * v;
    }
    const float scale = sqrtf(ns) + 1e-10f;
    const float fac   = (ns / (1.0f + ns)) / scale;
    #pragma unroll
    for (int c = 0; c < NCLS; ++c) {
        int k = (b * NCLS + c) * OUTC + o;
        out[k] = sj[k] * fac;
    }
}

// ---------------- b_ij[i,c] += mean_b sum_o u_j_i[b,i,c,o]*out[b,c,o] ----------------
__global__ void __launch_bounds__(256) uv_kernel(const float* __restrict__ uji,
                                                 const float* __restrict__ out,
                                                 float* __restrict__ bij) {
    __shared__ float red[256];
    const int ic = blockIdx.x;          // i*10 + c
    const int i  = ic / NCLS;
    const int c  = ic % NCLS;
    const int t  = threadIdx.x;
    const int o  = t & 15;
    const int b0 = t >> 4;

    float acc = 0.0f;
    for (int b = b0; b < BATCH; b += 16)
        acc += uji[(((size_t)b * IS + i) * NCLS + c) * OUTC + o] *
               out[(b * NCLS + c) * OUTC + o];
    red[t] = acc;
    __syncthreads();
    for (int s = 128; s > 0; s >>= 1) {
        if (t < s) red[t] += red[t + s];
        __syncthreads();
    }
    if (t == 0) bij[ic] += red[0] * (1.0f / (float)BATCH);
}

// ---------------- host-side orchestration ----------------
extern "C" void kernel_launch(void* const* d_in, const int* in_sizes, int n_in,
                              void* d_out, int out_size, void* d_ws, size_t ws_size,
                              hipStream_t stream) {
    const float* x       = (const float*)d_in[0];
    const float* conv_w  = (const float*)d_in[1];
    const float* conv_b  = (const float*)d_in[2];
    const float* prim_w  = (const float*)d_in[3];
    const float* prim_b  = (const float*)d_in[4];
    const float* digit_w = (const float*)d_in[5];
    float* outp = (float*)d_out;
    char*  ws   = (char*)d_ws;

    __bf16* xbf  = (__bf16*)(ws + OFF_XBF);
    __bf16* wcbf = (__bf16*)(ws + OFF_WCBF);
    __bf16* wpbf = (__bf16*)(ws + OFF_WPBF);
    __bf16* hbf  = (__bf16*)(ws + OFF_HBF);
    float*  p    = (float*) (ws + OFF_P);
    float*  ui   = (float*) (ws + OFF_UI);
    float*  uji  = (float*) (ws + OFF_UJI);
    float*  bij  = (float*) (ws + OFF_BIJ);
    float*  cij  = (float*) (ws + OFF_CIJ);
    float*  sj   = (float*) (ws + OFF_SJ);
    float*  outw = (float*) (ws + OFF_OUTW);

    cvt_bf16_kernel<<<(200704 + 255) / 256, 256, 0, stream>>>(x, xbf, 200704);
    cvt_bf16_kernel<<<(20736 + 255) / 256, 256, 0, stream>>>(conv_w, wcbf, 20736);
    cvt_bf16_kernel<<<(5308416 + 255) / 256, 256, 0, stream>>>(prim_w, wpbf, 5308416);

    conv1_wmma_kernel<<<(16 * 6400) / 8, 256, 0, stream>>>(xbf, wcbf, conv_b, hbf);
    conv2_wmma_kernel<<<16 * 18, 256, 0, stream>>>(wpbf, hbf, prim_b, p);

    squash_primary_kernel<<<BATCH * NCAP, 256, 0, stream>>>(p, ui);
    uji_kernel<<<(BATCH * IS * NCLS * OUTC) / 256, 256, 0, stream>>>(digit_w, ui, uji);

    zero_f32_kernel<<<(IS * NCLS + 255) / 256, 256, 0, stream>>>(bij, IS * NCLS);
    for (int it = 0; it < 3; ++it) {
        softmax_kernel<<<NCLS, 256, 0, stream>>>(bij, cij);
        sj_kernel<<<(BATCH * NCLS * OUTC + 255) / 256, 256, 0, stream>>>(cij, uji, sj);
        float* dst = (it == 2) ? outp : outw;
        squash_out_kernel<<<(BATCH * OUTC + 255) / 256, 256, 0, stream>>>(sj, dst);
        if (it < 2)
            uv_kernel<<<IS * NCLS, 256, 0, stream>>>(uji, dst, bij);
    }
}
